// Model_4398046511364
// MI455X (gfx1250) — matrix operand order repaired and checked by reading, not compiled, not run on hardware
//
#include <hip/hip_runtime.h>
#include <math.h>

#define N_GRAPHS 512
#define NODES_PER_GRAPH 1000
#define NTOT (N_GRAPHS * NODES_PER_GRAPH)   // 512000
#define EDGES (1 << 20)                     // 1048576
#define F_IN 256
#define NUM_CLASSES 50
#define K_POOL 30
#define F_CAT 97

typedef __attribute__((ext_vector_type(16))) __bf16 v16bf;
typedef __attribute__((ext_vector_type(8)))  float  v8f;

// ---------------- degree / D^{-1/2} ----------------
__global__ void k_init_deg(float* deg) {
  int i = blockIdx.x * blockDim.x + threadIdx.x;
  if (i < NTOT) deg[i] = 1.0f;  // self loop
}
__global__ void k_edge_deg(const int* __restrict__ dst, float* deg) {
  int e = blockIdx.x * blockDim.x + threadIdx.x;
  if (e < EDGES) atomicAdd(&deg[dst[e]], 1.0f);
}
__global__ void k_dinv(const float* __restrict__ deg, float* dinv) {
  int i = blockIdx.x * blockDim.x + threadIdx.x;
  if (i < NTOT) dinv[i] = rsqrtf(deg[i]);
}

// ---------------- WMMA GEMM: out[N,32] = A[:, off:off+F] @ W[F,32] ----------------
// 256 threads = 8 wave32; wave w owns rows [blk*128 + 16w, +16), 32 output cols
// as two 16x16 f32 accumulators; K stepped by 32 with bf16 v_wmma.
__global__ __launch_bounds__(256) void k_gemm_wmma(
    const float* __restrict__ A, int lda, int colOff, int F,
    const float* __restrict__ W, float* __restrict__ out) {
  __shared__ float sW[F_IN * 32];   // whole weight matrix (<= 32 KB)
  __shared__ float sA[128 * 32];    // one K-step tile (16 KB)
  const int tid  = threadIdx.x;
  const int lane = tid & 31;
  const int wv   = tid >> 5;
  const int rowBase = blockIdx.x * 128;

  for (int i = tid; i < F * 32; i += 256) sW[i] = W[i];

  v8f c0 = {}, c1 = {};
  const int nSteps = F >> 5;
  for (int kb = 0; kb < nSteps; ++kb) {
    __syncthreads();
    for (int i = tid; i < 128 * 32; i += 256) {
      int r = i >> 5, c = i & 31;
      sA[i] = A[(size_t)(rowBase + r) * lda + colOff + (kb << 5) + c];
    }
    if (kb + 1 < nSteps) {  // gfx1250 global_prefetch_b8 for next tile
      const float* nxt = &A[(size_t)(rowBase + (tid >> 1)) * lda + colOff +
                            ((kb + 1) << 5) + ((tid & 1) << 4)];
      __builtin_prefetch(nxt, 0, 1);
    }
    __syncthreads();

    // A fragment: lane = M (mod 16); lanes 16-31 hold upper 16 K (contiguous, matches B).
    v16bf a, b0, b1;
    const int mloc = (wv << 4) + (lane & 15);
    const int hi   = lane >> 4;
#pragma unroll
    for (int e = 0; e < 16; ++e) {
      int v = e >> 1, p = e & 1;
      int k = (hi << 4) + e;
      a[e] = (__bf16)sA[mloc * 32 + k];
    }
    // B fragment: lane = N (mod 16); lanes 16-31 hold upper 16 K rows.
    const int ncol = lane & 15;
    const int kAdd = (kb << 5) + (hi << 4);
#pragma unroll
    for (int e = 0; e < 16; ++e) {
      b0[e] = (__bf16)sW[(kAdd + e) * 32 + ncol];
      b1[e] = (__bf16)sW[(kAdd + e) * 32 + 16 + ncol];
    }
    c0 = __builtin_amdgcn_wmma_f32_16x16x32_bf16(false, a, false, b0, (short)0, c0, false, false);
    c1 = __builtin_amdgcn_wmma_f32_16x16x32_bf16(false, a, false, b1, (short)0, c1, false, false);
  }

  const int ncol = lane & 15;
#pragma unroll
  for (int r = 0; r < 8; ++r) {
    int m = r + ((lane >> 4) << 3);                 // C layout: VGPR r, half-wave
    size_t row = (size_t)rowBase + (wv << 4) + m;
    out[row * 32 + ncol]      = c0[r];
    out[row * 32 + 16 + ncol] = c1[r];
  }
}

// ---------------- GCN normalize/scatter/activate (32 ch) ----------------
__global__ void k_agg_init32(const float* __restrict__ xw, const float* __restrict__ dinv,
                             const float* __restrict__ b, float* agg) {
  int t = blockIdx.x * blockDim.x + threadIdx.x;
  if (t < NTOT * 32) {
    int i = t >> 5, c = t & 31;
    float di = dinv[i];
    agg[t] = di * di * xw[t] + b[c];
  }
}
__global__ void k_edge_scatter32(const float* __restrict__ xw, const float* __restrict__ dinv,
                                 const int* __restrict__ src, const int* __restrict__ dst,
                                 float* agg) {
  int t = blockIdx.x * blockDim.x + threadIdx.x;
  int e = t >> 5, c = t & 31;
  if (e < EDGES) {
    int s = src[e], d = dst[e];
    float w = dinv[s] * dinv[d];
    atomicAdd(&agg[(size_t)d * 32 + c], w * xw[(size_t)s * 32 + c]);
  }
}
__global__ void k_act32(const float* __restrict__ agg, float* feats, int off) {
  int t = blockIdx.x * blockDim.x + threadIdx.x;
  if (t < NTOT * 32) {
    int i = t >> 5, c = t & 31;
    feats[(size_t)i * F_CAT + off + c] = tanhf(agg[t]);
  }
}

// ---------------- layer 4: 32 -> 1 ----------------
__global__ void k_xw4(const float* __restrict__ feats, const float* __restrict__ W4, float* xw1) {
  int i = blockIdx.x * blockDim.x + threadIdx.x;
  if (i < NTOT) {
    float s = 0.0f;
#pragma unroll
    for (int c = 0; c < 32; ++c) s += feats[(size_t)i * F_CAT + 64 + c] * W4[c];
    xw1[i] = s;
  }
}
__global__ void k_agg_init1(const float* __restrict__ xw1, const float* __restrict__ dinv,
                            const float* __restrict__ b4, float* agg1) {
  int i = blockIdx.x * blockDim.x + threadIdx.x;
  if (i < NTOT) {
    float di = dinv[i];
    agg1[i] = di * di * xw1[i] + b4[0];
  }
}
__global__ void k_edge_scatter1(const float* __restrict__ xw1, const float* __restrict__ dinv,
                                const int* __restrict__ src, const int* __restrict__ dst,
                                float* agg1) {
  int e = blockIdx.x * blockDim.x + threadIdx.x;
  if (e < EDGES) {
    int s = src[e], d = dst[e];
    atomicAdd(&agg1[d], dinv[s] * dinv[d] * xw1[s]);
  }
}
__global__ void k_act1(const float* __restrict__ agg1, float* feats) {
  int i = blockIdx.x * blockDim.x + threadIdx.x;
  if (i < NTOT) feats[(size_t)i * F_CAT + 96] = tanhf(agg1[i]);
}

// ---------------- per-graph sort-pool + conv/FC head ----------------
__global__ __launch_bounds__(128) void k_head(
    const float* __restrict__ feats,
    const float* __restrict__ w5, const float* __restrict__ b5,
    const float* __restrict__ w6, const float* __restrict__ b6,
    const float* __restrict__ fc1w, const float* __restrict__ fc1b,
    const float* __restrict__ fc2w, const float* __restrict__ fc2b,
    float* __restrict__ out) {
  __shared__ float sv[NODES_PER_GRAPH];
  __shared__ int   sidx[K_POOL];
  __shared__ float stop[K_POOL * F_CAT];
  __shared__ float sh5[16 * 30];
  __shared__ float sm[16 * 15];
  __shared__ float sh6[32 * 11];
  __shared__ float sf1[128];
  __shared__ float slog[NUM_CLASSES];
  __shared__ float sred[2];

  const int g = blockIdx.x;
  const int tid = threadIdx.x;
  const int bd = blockDim.x;
  const float* fg = feats + (size_t)g * NODES_PER_GRAPH * F_CAT;

  for (int j = tid; j < NODES_PER_GRAPH; j += bd) sv[j] = fg[(size_t)j * F_CAT + 96];
  __syncthreads();

  // rank = #{greater} + #{equal with smaller index}  == stable argsort(-v)
  for (int j = tid; j < NODES_PER_GRAPH; j += bd) {
    float v = sv[j];
    int r = 0;
    for (int t = 0; t < NODES_PER_GRAPH; ++t) {
      float u = sv[t];
      r += (u > v) || (u == v && t < j);
    }
    if (r < K_POOL) sidx[r] = j;
  }
  __syncthreads();

  for (int i = tid; i < K_POOL * F_CAT; i += bd) {
    int k = i / F_CAT, c = i - k * F_CAT;
    stop[i] = fg[(size_t)sidx[k] * F_CAT + c];
  }
  __syncthreads();

  // conv5: kernel 97 / stride 97 == per pooled node, 97->16 linear + ReLU
  for (int i = tid; i < 16 * 30; i += bd) {
    int c = i / 30, k = i - c * 30;
    float s = b5[c];
    for (int q = 0; q < F_CAT; ++q) s += w5[c * F_CAT + q] * stop[k * F_CAT + q];
    sh5[i] = fmaxf(s, 0.0f);
  }
  __syncthreads();

  for (int i = tid; i < 16 * 15; i += bd) {
    int c = i / 15, t = i - c * 15;
    sm[i] = fmaxf(sh5[c * 30 + 2 * t], sh5[c * 30 + 2 * t + 1]);
  }
  __syncthreads();

  for (int i = tid; i < 32 * 11; i += bd) {
    int c = i / 11, t = i - c * 11;
    float s = b6[c];
    for (int ci = 0; ci < 16; ++ci)
#pragma unroll
      for (int q = 0; q < 5; ++q)
        s += w6[(c * 16 + ci) * 5 + q] * sm[ci * 15 + t + q];
    sh6[i] = fmaxf(s, 0.0f);
  }
  __syncthreads();

  for (int o = tid; o < 128; o += bd) {
    float s = fc1b[o];
    for (int j = 0; j < 352; ++j) s += sh6[j] * fc1w[j * 128 + o];
    sf1[o] = fmaxf(s, 0.0f);
  }
  __syncthreads();

  for (int o = tid; o < NUM_CLASSES; o += bd) {
    float s = fc2b[o];
    for (int j = 0; j < 128; ++j) s += sf1[j] * fc2w[j * NUM_CLASSES + o];
    slog[o] = s;
  }
  __syncthreads();

  if (tid == 0) {
    float m = slog[0];
    for (int o = 1; o < NUM_CLASSES; ++o) m = fmaxf(m, slog[o]);
    float se = 0.0f;
    for (int o = 0; o < NUM_CLASSES; ++o) se += expf(slog[o] - m);
    sred[0] = m;
    sred[1] = logf(se);
  }
  __syncthreads();
  for (int o = tid; o < NUM_CLASSES; o += bd)
    out[(size_t)g * NUM_CLASSES + o] = slog[o] - sred[0] - sred[1];
}

// ---------------- orchestration ----------------
extern "C" void kernel_launch(void* const* d_in, const int* in_sizes, int n_in,
                              void* d_out, int out_size, void* d_ws, size_t ws_size,
                              hipStream_t stream) {
  (void)in_sizes; (void)n_in; (void)out_size; (void)ws_size;
  const float* x   = (const float*)d_in[0];
  const int*   ei  = (const int*)d_in[1];
  const int*   src = ei;
  const int*   dst = ei + EDGES;
  const float* W1 = (const float*)d_in[3];
  const float* b1 = (const float*)d_in[4];
  const float* W2 = (const float*)d_in[5];
  const float* b2 = (const float*)d_in[6];
  const float* W3 = (const float*)d_in[7];
  const float* b3 = (const float*)d_in[8];
  const float* W4 = (const float*)d_in[9];
  const float* b4 = (const float*)d_in[10];
  const float* w5 = (const float*)d_in[11];
  const float* b5 = (const float*)d_in[12];
  const float* w6 = (const float*)d_in[13];
  const float* b6 = (const float*)d_in[14];
  const float* fc1w = (const float*)d_in[15];
  const float* fc1b = (const float*)d_in[16];
  const float* fc2w = (const float*)d_in[17];
  const float* fc2b = (const float*)d_in[18];
  float* out = (float*)d_out;

  // workspace: dinv[N] | xw[N*32] | agg[N*32] | feats[N*97]  (~332 MB)
  float* ws    = (float*)d_ws;
  float* dinv  = ws;
  float* xw    = dinv + NTOT;
  float* agg   = xw + (size_t)NTOT * 32;
  float* feats = agg + (size_t)NTOT * 32;
  float* deg   = xw;  // transient reuse before first GEMM

  const int TPB = 256;
  k_init_deg<<<NTOT / TPB, TPB, 0, stream>>>(deg);
  k_edge_deg<<<EDGES / TPB, TPB, 0, stream>>>(dst, deg);
  k_dinv<<<NTOT / TPB, TPB, 0, stream>>>(deg, dinv);

  struct Layer { const float* A; int lda; int off; int F; const float* W; const float* b; int fo; };
  Layer L[3] = {
    { x,     F_IN,  0,  F_IN, W1, b1, 0  },
    { feats, F_CAT, 0,  32,   W2, b2, 32 },
    { feats, F_CAT, 32, 32,   W3, b3, 64 },
  };
  for (int l = 0; l < 3; ++l) {
    k_gemm_wmma<<<NTOT / 128, 256, 0, stream>>>(L[l].A, L[l].lda, L[l].off, L[l].F, L[l].W, xw);
    k_agg_init32<<<NTOT * 32 / TPB, TPB, 0, stream>>>(xw, dinv, L[l].b, agg);
    k_edge_scatter32<<<EDGES * 32 / TPB, TPB, 0, stream>>>(xw, dinv, src, dst, agg);
    k_act32<<<NTOT * 32 / TPB, TPB, 0, stream>>>(agg, feats, L[l].fo);
  }

  // layer 4: 32 -> 1
  k_xw4<<<NTOT / TPB, TPB, 0, stream>>>(feats, W4, xw);
  k_agg_init1<<<NTOT / TPB, TPB, 0, stream>>>(xw, dinv, b4, agg);
  k_edge_scatter1<<<EDGES / TPB, TPB, 0, stream>>>(xw, dinv, src, dst, agg);
  k_act1<<<NTOT / TPB, TPB, 0, stream>>>(agg, feats);

  k_head<<<N_GRAPHS, 128, 0, stream>>>(feats, w5, b5, w6, b6, fc1w, fc1b, fc2w, fc2b, out);
}
